// Decoder_BCD_43138651521373
// MI455X (gfx1250) — matrix operand order repaired and checked by reading, not compiled, not run on hardware
//
#include <hip/hip_runtime.h>
#include <hip/hip_bf16.h>

// ---------------- types ----------------
typedef _Float16 h16;
typedef __attribute__((ext_vector_type(16))) _Float16 v16h;
typedef __attribute__((ext_vector_type(8)))  _Float16 v8h;
typedef __attribute__((ext_vector_type(8)))  float    v8f;

union Frag16 { v16h v; v8h h[2]; };

#define DIMN 64
#define LDIM 2016
#define DD   2017   // L_DIM + NOISE_DIM
#define DPAD 2048
#define BB   512
#define HIDN 512
#define NOUT 4096   // DIM*DIM
#define NINT 512

// output offsets (floats)
#define OFF_P 0
#define OFF_L (BB*DIMN*DIMN)                 // 2097152
#define OFF_N (OFF_L + BB*DIMN*DIMN)         // 4194304
#define OFF_W (OFF_N + BB*DIMN)              // 4227072
#define OFF_S (OFF_W + BB*DIMN*DIMN)         // 6324224

// ---------------- prep: full_l_batch (f32 + padded f16), w_noise ----------------
__global__ void prep_kernel(const float* __restrict__ Lp, const float* __restrict__ epsL,
                            float* __restrict__ fl32, h16* __restrict__ fl16,
                            float* __restrict__ wn) {
  int idx = blockIdx.x * 256 + threadIdx.x;
  int b = idx >> 11, d = idx & (DPAD - 1);
  if (b >= BB) return;
  if (d < DD) {
    float mean = Lp[d];
    float raw  = Lp[DD + d];
    float ls   = tanhf(raw * 0.1f) * 10.0f;
    float v    = mean + expf(ls) * epsL[b * DD + d];
    fl32[b * DD + d]   = v;
    fl16[b * DPAD + d] = (h16)v;
    if (d == LDIM) wn[b] = v;      // noise column
  } else {
    fl16[b * DPAD + d] = (h16)0.0f;
  }
}

// ---------------- batched_noises output ----------------
__global__ void noises_kernel(const float* __restrict__ wn, float* __restrict__ outN) {
  int idx = blockIdx.x * 256 + threadIdx.x;
  if (idx < BB * DIMN) outN[idx] = wn[idx >> 6];
}

// ---------------- batched_L output: strictly lower triangular scatter ----------------
__global__ void lowerL_kernel(const float* __restrict__ fl32, float* __restrict__ outL) {
  int idx = blockIdx.x * 256 + threadIdx.x;   // over B*64*64
  int b  = idx >> 12;
  int rc = idx & 4095;
  int rr = rc >> 6, cc = rc & 63;
  if (b >= BB) return;
  float v = 0.0f;
  if (rr > cc) {
    // triu index for (i=cc, j=rr): t = 63*i - i*(i-1)/2 + (j - i - 1)
    int t = 63 * cc - (cc * (cc - 1)) / 2 + (rr - cc - 1);
    v = fl32[b * DD + t];
  }
  outL[idx] = v;
}

// ---------------- weight transpose + f32->f16 (dst[n][k], padded K) ----------------
__global__ void convT_kernel(const float* __restrict__ src, h16* __restrict__ dst,
                             int K, int N, int Kp) {
  long idx = (long)blockIdx.x * 256 + threadIdx.x;
  if (idx >= (long)N * Kp) return;
  int n = (int)(idx / Kp), k = (int)(idx % Kp);
  dst[idx] = (k < K) ? (h16)src[(long)k * N + n] : (h16)0.0f;
}

// ---------------- WMMA GEMM: C[m,n] = act( sum_k A[m,k]*Bt[n,k] + bias[n] ) -------
// Each wave computes a 16(M) x 64(N) strip: one A fragment feeds 4 WMMAs per K-step.
// Compile-time K + explicit double buffering -> loads for step k+1 are in flight
// while the 4 WMMAs of step k execute (no full s_wait_loadcnt stalls).
// ACT==0: gelu(tanh approx) -> f16 out ; ACT==1: tanh(x/10)*10 -> f32 out
template<int ACT, int K, int LDA, int LDB, int LDO>
__global__ void wmma_gemm_kernel(const h16* __restrict__ A, const h16* __restrict__ Bt,
                                 const float* __restrict__ bias,
                                 h16* __restrict__ out16, float* __restrict__ out32) {
  const int lane = threadIdx.x & 31;
  const int wave = threadIdx.x >> 5;
  const int lm = lane & 15, hi = lane >> 4;
  const int m0 = blockIdx.x * 16;
  const int nW = blockIdx.y * 256 + wave * 64;

  const h16* arow = A + (long)(m0 + lm) * LDA;
  const h16* brow[4];
#pragma unroll
  for (int f = 0; f < 4; ++f) brow[f] = Bt + (long)(nW + f * 16 + lm) * LDB;

  // A 16x32 f16 fragment: K in [16*hi, 16*hi+15]
  auto loadA = [&](Frag16& a, int kk) {
    a.h[0] = *(const v8h*)(arow + kk + 16 * hi);
    a.h[1] = *(const v8h*)(arow + kk + 16 * hi + 8);
  };
  // B 32x16 f16 fragment (Bt is N-major): K in [16*hi, 16*hi+15], N = lm
  auto loadB = [&](Frag16* bf, int kk) {
#pragma unroll
    for (int f = 0; f < 4; ++f) {
      bf[f].h[0] = *(const v8h*)(brow[f] + kk + 16 * hi);
      bf[f].h[1] = *(const v8h*)(brow[f] + kk + 16 * hi + 8);
    }
  };

  Frag16 aC, bC[4], aN, bN[4];
  loadA(aC, 0);
  loadB(bC, 0);
  v8f acc[4] = {};
#pragma unroll
  for (int kk = 0; kk < K; kk += 32) {
    if (kk + 32 < K) {
      __builtin_prefetch(arow + kk + 160, 0, 1);   // global_prefetch_b8
      loadA(aN, kk + 32);
      loadB(bN, kk + 32);
    }
#pragma unroll
    for (int f = 0; f < 4; ++f)
      acc[f] = __builtin_amdgcn_wmma_f32_16x16x32_f16(false, aC.v, false, bC[f].v,
                                                      (short)0, acc[f], false, false);
    aC = aN;
#pragma unroll
    for (int f = 0; f < 4; ++f) bC[f] = bN[f];
  }

#pragma unroll
  for (int f = 0; f < 4; ++f) {
    const int n = nW + f * 16 + lm;
    const float bn = bias[n];
#pragma unroll
    for (int r = 0; r < 8; ++r) {
      int m = m0 + r + 8 * hi;       // C layout: VGPR r -> M = r + 8*(lane>>4)
      float x = acc[f][r] + bn;
      if (ACT == 0) {
        float x3 = x * x * x;
        float y  = 0.5f * x * (1.0f + tanhf(0.7978845608028654f * (x + 0.044715f * x3)));
        out16[(long)m * LDO + n] = (h16)y;
      } else {
        float y = tanhf(x * 0.1f) * 10.0f;
        out32[(long)m * LDO + n] = y;
      }
    }
  }
}

// ---------------- sinkhorn + greedy perm + P/W outputs ----------------
__global__ void sinkhorn_kernel(const float* __restrict__ logits, const float* __restrict__ gumbel,
                                const float* __restrict__ outL,
                                float* __restrict__ outP, float* __restrict__ outW,
                                int* __restrict__ cols, int* __restrict__ pinv) {
  __shared__ float tile[64 * 65];
  __shared__ float colLse[64];
  __shared__ int js[64];
  __shared__ int msk[64];
  const int b = blockIdx.x;
  const int t = threadIdx.x;   // row owner, blockDim = 64

  float r[64];
#pragma unroll
  for (int c = 0; c < 64; ++c) {
    float u = gumbel[b * NOUT + t * 64 + c];
    u = fminf(fmaxf(u, 1e-8f), 1.0f - 1e-8f);
    float g = -logf(-logf(u));
    r[c] = (logits[b * NOUT + t * 64 + c] + g) * 5.0f;   // / TAU (0.2)
  }

  for (int it = 0; it < 20; ++it) {
    // row logsumexp (registers)
    float mx = -3.4e38f;
#pragma unroll
    for (int c = 0; c < 64; ++c) mx = fmaxf(mx, r[c]);
    float s = 0.0f;
#pragma unroll
    for (int c = 0; c < 64; ++c) s += expf(r[c] - mx);
    float lse = mx + logf(s);
#pragma unroll
    for (int c = 0; c < 64; ++c) { r[c] -= lse; tile[t * 65 + c] = r[c]; }
    __syncthreads();
    // column logsumexp (thread t owns column t)
    float cm = -3.4e38f;
    for (int i = 0; i < 64; ++i) cm = fmaxf(cm, tile[i * 65 + t]);
    float cs = 0.0f;
    for (int i = 0; i < 64; ++i) cs += expf(tile[i * 65 + t] - cm);
    colLse[t] = cm + logf(cs);
    __syncthreads();
#pragma unroll
    for (int c = 0; c < 64; ++c) r[c] -= colLse[c];
    __syncthreads();
  }

  // greedy permutation over rows 0..63 with column mask (argmax of masked
  // log-values == argmax of masked soft, exp is monotonic)
  msk[t] = 1;
  __syncthreads();
  for (int i = 0; i < 64; ++i) {
    if (t == i) {
      float best = -3.4e38f; int jb = 0;
#pragma unroll
      for (int c = 0; c < 64; ++c) {
        float v = msk[c] ? r[c] : -3.4e38f;
        if (v > best) { best = v; jb = c; }
      }
      js[i] = jb; msk[jb] = 0;
    }
    __syncthreads();
  }

  const int jt = js[t];
  cols[b * 64 + t]  = jt;
  pinv[b * 64 + jt] = t;
  // batched_P forward value == hard one-hot
#pragma unroll
  for (int c = 0; c < 64; ++c) outP[b * NOUT + t * 64 + c] = (c == jt) ? 1.0f : 0.0f;
  // batched_W = P L P^T  ==  W[r,c] = L[js[r], js[c]]
  const float* Lb = outL + (long)b * NOUT;
#pragma unroll
  for (int c = 0; c < 64; ++c) outW[b * NOUT + t * 64 + c] = Lb[jt * 64 + js[c]];
}

// ---------------- ancestral sampling: triangular backward substitution -----------
// L_b staged to LDS with gfx1250 async memory->LDS loads (ASYNCcnt path).
__global__ void ancestral_kernel(const float* __restrict__ outL, const float* __restrict__ z,
                                 const int* __restrict__ targ,
                                 const int* __restrict__ cols, const int* __restrict__ pinv,
                                 const float* __restrict__ wn, float* __restrict__ samples) {
  __shared__ float Ls[64 * 64];
  __shared__ int jss[64];
  __shared__ int ps[64];
  const int b = blockIdx.x;
  const int tid = threadIdx.x;            // blockDim = 256

  // ---- async copy of the 16KB L matrix: 256 lanes x 4 iters x b128 ----
  // generic pointers to LDS carry the DS byte-offset in the low 32 bits
  unsigned ldsbase = (unsigned)(unsigned long long)(&Ls[0]);
  const float* gsrc = outL + (long)b * NOUT;
#pragma unroll
  for (int it = 0; it < 4; ++it) {
    int foff = (it * 256 + tid) * 4;              // float index, 16B per lane
    unsigned laddr = ldsbase + (unsigned)(foff * 4);
    const float* gaddr = gsrc + foff;
    asm volatile("global_load_async_to_lds_b128 %0, %1, off"
                 :: "v"(laddr), "v"(gaddr) : "memory");
  }
  if (tid < 64) { jss[tid] = cols[b * 64 + tid]; ps[tid] = pinv[b * 64 + tid]; }
  asm volatile("s_wait_asynccnt 0x0" ::: "memory");
  __syncthreads();

  const int i = blockIdx.y * 256 + tid;   // intervention sample
  const float w = wn[b];
  const int ti = targ[i];
  const int mt = (ti < 64) ? jss[ti] : 127;   // clamped node in permuted coords

  // u[m] = noise_p[m] + sum_{m' > m} u[m'] * L[m', m], m = 63 .. 0
  float u[64];
#pragma unroll
  for (int m = 63; m >= 0; --m) {
    float acc = w * z[i * 64 + ps[m]];
#pragma unroll
    for (int m2 = m + 1; m2 < 64; ++m2)
      acc = fmaf(u[m2], Ls[m2 * 64 + m], acc);   // LDS broadcast (uniform addr)
    u[m] = (m == mt) ? 0.0f : acc;
  }

  float* so = samples + ((long)b * NINT + i) * 64;
#pragma unroll
  for (int m = 0; m < 64; ++m) so[ps[m]] = u[m];   // s[row] = u[js[row]]
}

// ---------------- launcher ----------------
extern "C" void kernel_launch(void* const* d_in, const int* in_sizes, int n_in,
                              void* d_out, int out_size, void* d_ws, size_t ws_size,
                              hipStream_t stream) {
  (void)in_sizes; (void)n_in; (void)out_size; (void)ws_size;
  const float* Lp    = (const float*)d_in[0];
  const float* epsL  = (const float*)d_in[1];
  const float* gumb  = (const float*)d_in[2];
  const float* znoi  = (const float*)d_in[3];
  const float* pW1   = (const float*)d_in[4];
  const float* pb1   = (const float*)d_in[5];
  const float* pW2   = (const float*)d_in[6];
  const float* pb2   = (const float*)d_in[7];
  const float* pW3   = (const float*)d_in[8];
  const float* pb3   = (const float*)d_in[9];
  const int*   targ  = (const int*)d_in[10];
  float* out = (float*)d_out;

  // workspace carve (256B aligned slabs)
  char* p = (char*)d_ws;
  auto take = [&](size_t bytes) { char* q = p; p += (bytes + 255) & ~(size_t)255; return q; };
  float* fl32   = (float*)take((size_t)BB * DD * 4);
  h16*   fl16   = (h16*)  take((size_t)BB * DPAD * 2);
  h16*   W1t    = (h16*)  take((size_t)HIDN * DPAD * 2);
  h16*   W2t    = (h16*)  take((size_t)HIDN * HIDN * 2);
  h16*   W3t    = (h16*)  take((size_t)NOUT * HIDN * 2);
  h16*   h1     = (h16*)  take((size_t)BB * HIDN * 2);
  h16*   h2     = (h16*)  take((size_t)BB * HIDN * 2);
  float* logit  = (float*)take((size_t)BB * NOUT * 4);
  float* wnbuf  = (float*)take((size_t)BB * 4);
  int*   colsWS = (int*)  take((size_t)BB * 64 * 4);
  int*   pinvWS = (int*)  take((size_t)BB * 64 * 4);

  float* outP = out + OFF_P;
  float* outL = out + OFF_L;
  float* outN = out + OFF_N;
  float* outW = out + OFF_W;
  float* outS = out + OFF_S;

  // 1. elementwise prep
  prep_kernel<<<(BB * DPAD) / 256, 256, 0, stream>>>(Lp, epsL, fl32, fl16, wnbuf);
  noises_kernel<<<(BB * DIMN) / 256, 256, 0, stream>>>(wnbuf, outN);
  lowerL_kernel<<<(BB * NOUT) / 256, 256, 0, stream>>>(fl32, outL);

  // 2. weight staging (transpose to N-major fp16, pad K)
  convT_kernel<<<((long)HIDN * DPAD + 255) / 256, 256, 0, stream>>>(pW1, W1t, DD, HIDN, DPAD);
  convT_kernel<<<((long)HIDN * HIDN + 255) / 256, 256, 0, stream>>>(pW2, W2t, HIDN, HIDN, HIDN);
  convT_kernel<<<((long)NOUT * HIDN + 255) / 256, 256, 0, stream>>>(pW3, W3t, HIDN, NOUT, HIDN);

  // 3. MLP: WMMA GEMMs (128 threads = 4 waves; each wave a 16x64 strip)
  wmma_gemm_kernel<0, DPAD, DPAD, DPAD, HIDN>
      <<<dim3(BB / 16, HIDN / 256), 128, 0, stream>>>(fl16, W1t, pb1, h1, nullptr);
  wmma_gemm_kernel<0, HIDN, HIDN, HIDN, HIDN>
      <<<dim3(BB / 16, HIDN / 256), 128, 0, stream>>>(h1, W2t, pb2, h2, nullptr);
  wmma_gemm_kernel<1, HIDN, HIDN, HIDN, NOUT>
      <<<dim3(BB / 16, NOUT / 256), 128, 0, stream>>>(h2, W3t, pb3, nullptr, logit);

  // 4. sinkhorn + greedy -> P, W, perm tables
  sinkhorn_kernel<<<BB, 64, 0, stream>>>(logit, gumb, outL, outP, outW, colsWS, pinvWS);

  // 5. ancestral sampling
  ancestral_kernel<<<dim3(BB, NINT / 256), 256, 0, stream>>>(
      outL, znoi, targ, colsWS, pinvWS, wnbuf, outS);
}
